// Qwen3VisionTransformer_19756849561611
// MI455X (gfx1250) — matrix operand order repaired and checked by reading, not compiled, not run on hardware
//
#include <hip/hip_runtime.h>
#include <stdint.h>

typedef __bf16 bf16_t;
typedef __attribute__((ext_vector_type(16))) __bf16 v16bf;
typedef __attribute__((ext_vector_type(8)))  float  v8f;

#define NTOK  16384
#define HID   1280
#define QKV3  3840
#define INTER 3456
#define PIX   1536
#define NHEAD 16
#define HD    80
#define NWIN  256
#define MROW  4096
#define UHID  5120
#define ODIM  2048

__device__ __forceinline__ bf16_t f2bf(float f) {
  union { float f; uint32_t u; } v; v.f = f;
  uint32_t r = v.u + 0x7FFFu + ((v.u >> 16) & 1u);
  uint16_t h = (uint16_t)(r >> 16);
  bf16_t b; __builtin_memcpy(&b, &h, 2); return b;
}
__device__ __forceinline__ bf16_t bfzero() {
  uint16_t h = 0; bf16_t b; __builtin_memcpy(&b, &h, 2); return b;
}
__device__ __forceinline__ float gelu_tanh(float x) {
  return 0.5f * x * (1.f + tanhf(0.7978845608028654f * (x + 0.044715f * x * x * x)));
}

// ---------------------------------------------------------------------------
// CDNA5 async global->LDS copy (ASYNCcnt path, 16B per lane per issue).
// Tile: 128 rows x 64 bf16 cols, LDS row stride 72 bf16.
// ---------------------------------------------------------------------------
__device__ __forceinline__ void issue_tile_async(const bf16_t* gsrc, int ldg,
                                                 bf16_t* lbase, int tid)
{
  #pragma unroll
  for (int i = 0; i < 4; ++i) {
    int c = tid + i * 256;                // 1024 chunks of 16B
    int r = c >> 3, col = (c & 7) * 8;
    uint32_t voff  = (uint32_t)(((size_t)r * ldg + col) * 2);
    uint32_t laddr = (uint32_t)(uintptr_t)(lbase + r * 72 + col);
    asm volatile("global_load_async_to_lds_b128 %0, %1, %2"
                 :: "v"(laddr), "v"(voff), "s"((uint64_t)(uintptr_t)gsrc)
                 : "memory");
  }
}
__device__ __forceinline__ void wait_async_all() {
  asm volatile("s_wait_asynccnt 0x0" ::: "memory");
}

// ---------------------------------------------------------------------------
// General bf16 WMMA GEMM:  out[M,N] = A[M,K] (bf16, row-major) * Bt[N,K]^T + bias
// Block tile 128x128x64, double-buffered async LDS staging.
// 8 waves as 2(M) x 4(N); wave tile 64x32 -> 8 WMMAs / K-step.
// EPI 0: fp32 store   1: fp32 accumulate (residual)   2: gelu -> bf16
// EPI 3: fp32 store with row scatter via scat[]
// ---------------------------------------------------------------------------
template<int EPI>
__global__ __launch_bounds__(256, 2)
void gemm_bf16_wmma(const bf16_t* __restrict__ A, const bf16_t* __restrict__ Bt,
                    const float* __restrict__ bias,
                    float* __restrict__ outF, bf16_t* __restrict__ outB,
                    const int* __restrict__ scat,
                    int M, int N, int K)
{
  constexpr int BM = 128, BN = 128, BK = 64, LD = BK + 8;
  extern __shared__ char smem[];
  bf16_t* sA = (bf16_t*)smem;                          // 2 x BM*LD
  bf16_t* sB = (bf16_t*)(smem + 2 * BM * LD * 2);      // 2 x BN*LD
  const int bm = blockIdx.y * BM;
  const int bn = blockIdx.x * BN;
  const int tid  = threadIdx.x;
  const int wave = tid >> 5, lane = tid & 31;
  const int hf = lane >> 4, l16 = lane & 15;
  const int wm = (wave >> 2) * 64;   // 2 wave rows along M
  const int wn = (wave & 3) * 32;    // 4 wave cols along N

  v8f acc[4][2] = {};

  const bf16_t* ga = A + (size_t)bm * K;
  const bf16_t* gb = Bt + (size_t)bn * K;

  issue_tile_async(ga, K, sA, tid);
  issue_tile_async(gb, K, sB, tid);

  int buf = 0;
  for (int k0 = 0; k0 < K; k0 += BK) {
    wait_async_all();
    __syncthreads();
    const int nxt = buf ^ 1;
    if (k0 + BK < K) {
      issue_tile_async(ga + k0 + BK, K, sA + nxt * BM * LD, tid);
      issue_tile_async(gb + k0 + BK, K, sB + nxt * BN * LD, tid);
    }
    const bf16_t* cA = sA + buf * BM * LD;
    const bf16_t* cB = sB + buf * BN * LD;
    #pragma unroll
    for (int kk = 0; kk < BK; kk += 32) {
      v16bf af[4], bfv[2];
      #pragma unroll
      for (int t = 0; t < 4; ++t) {
        union { v16bf v; bf16_t e[16]; } ua;
        const bf16_t* pa = &cA[(wm + t * 16 + l16) * LD + kk + hf * 8];
        __builtin_memcpy(&ua.e[0], pa, 16);
        __builtin_memcpy(&ua.e[8], pa + 16, 16);
        af[t] = ua.v;
      }
      #pragma unroll
      for (int t = 0; t < 2; ++t) {
        union { v16bf v; bf16_t e[16]; } ub;
        const bf16_t* pb = &cB[(wn + t * 16 + l16) * LD + kk + hf * 8];
        __builtin_memcpy(&ub.e[0], pb, 16);
        __builtin_memcpy(&ub.e[8], pb + 16, 16);
        bfv[t] = ub.v;
      }
      #pragma unroll
      for (int i = 0; i < 4; ++i)
        #pragma unroll
        for (int j = 0; j < 2; ++j)
          acc[i][j] = __builtin_amdgcn_wmma_f32_16x16x32_bf16(
              false, af[i], false, bfv[j], (short)0, acc[i][j], false, false);
    }
    __syncthreads();
    buf = nxt;
  }

  #pragma unroll
  for (int j = 0; j < 2; ++j) {
    int n = bn + wn + j * 16 + l16;
    float bv = bias ? bias[n] : 0.f;
    #pragma unroll
    for (int i = 0; i < 4; ++i) {
      #pragma unroll
      for (int r = 0; r < 8; ++r) {
        int m = bm + wm + i * 16 + hf * 8 + r;
        float v = acc[i][j][r] + bv;
        if constexpr (EPI == 0)      { outF[(size_t)m * N + n] = v; }
        else if constexpr (EPI == 1) { outF[(size_t)m * N + n] += v; }
        else if constexpr (EPI == 2) { outB[(size_t)m * N + n] = f2bf(gelu_tanh(v)); }
        else                         { outF[(size_t)scat[m] * N + n] = v; }
      }
    }
  }
}
#define GEMM_SMEM ((2 * 128 * 72 + 2 * 128 * 72) * 2)

// ---------------------------------------------------------------------------
// Windowed attention: one block per (window, head), 4 waves.
// S = Q K^T / sqrt(80) -> softmax -> O = P V, both via WMMA.
// ---------------------------------------------------------------------------
__global__ __launch_bounds__(128, 2)
void attn_wmma(const bf16_t* __restrict__ qg, const bf16_t* __restrict__ kg,
               const bf16_t* __restrict__ vg, bf16_t* __restrict__ og)
{
  constexpr int LQ = 96 + 8, LV = 64 + 8, LP = 64 + 8;
  __shared__ bf16_t sQ[64 * LQ];
  __shared__ bf16_t sK[64 * LQ];
  __shared__ bf16_t sVt[80 * LV];
  __shared__ bf16_t sP[64 * LP];
  const int w = blockIdx.x >> 4, hh = blockIdx.x & 15;
  const int tid = threadIdx.x, wave = tid >> 5, lane = tid & 31;
  const int hf = lane >> 4, l16 = lane & 15;
  const size_t gbase = (size_t)w * 64 * HID + hh * HD;

  for (int i = tid; i < 64 * 96; i += 128) {
    int r = i / 96, c = i % 96;
    bf16_t qv = bfzero(), kv = bfzero();
    if (c < HD) { qv = qg[gbase + (size_t)r * HID + c];
                  kv = kg[gbase + (size_t)r * HID + c]; }
    sQ[r * LQ + c] = qv;
    sK[r * LQ + c] = kv;
  }
  for (int i = tid; i < 64 * HD; i += 128) {   // V transposed: sVt[d][token]
    int r = i / HD, c = i % HD;
    sVt[c * LV + r] = vg[gbase + (size_t)r * HID + c];
  }
  __syncthreads();

  const int mb = wave * 16;                     // 16-row strip per wave
  v8f sc[4] = {};
  #pragma unroll
  for (int kk = 0; kk < 96; kk += 32) {
    union { v16bf v; bf16_t e[16]; } ua;
    const bf16_t* pa = &sQ[(mb + l16) * LQ + kk + hf * 8];
    __builtin_memcpy(&ua.e[0], pa, 16);
    __builtin_memcpy(&ua.e[8], pa + 16, 16);
    #pragma unroll
    for (int t = 0; t < 4; ++t) {
      union { v16bf v; bf16_t e[16]; } ub;
      const bf16_t* pb = &sK[(t * 16 + l16) * LQ + kk + hf * 8];
      __builtin_memcpy(&ub.e[0], pb, 16);
      __builtin_memcpy(&ub.e[8], pb + 16, 16);
      sc[t] = __builtin_amdgcn_wmma_f32_16x16x32_bf16(
          false, ua.v, false, ub.v, (short)0, sc[t], false, false);
    }
  }
  const float scl = 0.11180339887498949f;       // 80^-0.5
  #pragma unroll
  for (int j = 0; j < 8; ++j) {
    float mx = -3.4e38f;
    #pragma unroll
    for (int t = 0; t < 4; ++t) mx = fmaxf(mx, sc[t][j] * scl);
    #pragma unroll
    for (int d = 1; d < 16; d <<= 1) mx = fmaxf(mx, __shfl_xor(mx, d, 32));
    float ev[4], sum = 0.f;
    #pragma unroll
    for (int t = 0; t < 4; ++t) { ev[t] = __expf(sc[t][j] * scl - mx); sum += ev[t]; }
    #pragma unroll
    for (int d = 1; d < 16; d <<= 1) sum += __shfl_xor(sum, d, 32);
    float inv = 1.f / sum;
    int m = mb + hf * 8 + j;
    #pragma unroll
    for (int t = 0; t < 4; ++t) sP[m * LP + t * 16 + l16] = f2bf(ev[t] * inv);
  }
  __syncthreads();

  v8f oc[5] = {};
  #pragma unroll
  for (int kk = 0; kk < 64; kk += 32) {
    union { v16bf v; bf16_t e[16]; } ua;
    const bf16_t* pa = &sP[(mb + l16) * LP + kk + hf * 8];
    __builtin_memcpy(&ua.e[0], pa, 16);
    __builtin_memcpy(&ua.e[8], pa + 16, 16);
    #pragma unroll
    for (int t = 0; t < 5; ++t) {
      union { v16bf v; bf16_t e[16]; } ub;
      const bf16_t* pb = &sVt[(t * 16 + l16) * LV + kk + hf * 8];
      __builtin_memcpy(&ub.e[0], pb, 16);
      __builtin_memcpy(&ub.e[8], pb + 16, 16);
      oc[t] = __builtin_amdgcn_wmma_f32_16x16x32_bf16(
          false, ua.v, false, ub.v, (short)0, oc[t], false, false);
    }
  }
  #pragma unroll
  for (int t = 0; t < 5; ++t)
    #pragma unroll
    for (int j = 0; j < 8; ++j) {
      int m = mb + hf * 8 + j, n = t * 16 + l16;
      og[gbase + (size_t)m * HID + n] = f2bf(oc[t][j]);
    }
}

// ---------------------------------------------------------------------------
// Support kernels
// ---------------------------------------------------------------------------
__global__ void cvtw_kernel(const float* __restrict__ w, bf16_t* __restrict__ wt,
                            int K, int N)   // fp32 [K][N] -> bf16 [N][K]
{
  long long i = (long long)blockIdx.x * 256 + threadIdx.x;
  if (i >= (long long)K * N) return;
  int k = (int)(i / N), n = (int)(i % N);
  wt[(size_t)n * K + k] = f2bf(w[i]);
}

__global__ void gather_px_kernel(const float* __restrict__ px,
                                 const int* __restrict__ widx,
                                 bf16_t* __restrict__ out)
{
  int i = blockIdx.x * 256 + threadIdx.x;     // NTOK*PIX, exact multiple
  int r = i / PIX, c = i % PIX;
  int src = widx[r >> 2] * 4 + (r & 3);
  out[i] = f2bf(px[(size_t)src * PIX + c]);
}

__global__ __launch_bounds__(256)
void rmsnorm_kernel(const float* __restrict__ x, const float* __restrict__ w,
                    bf16_t* __restrict__ y, int W)
{
  __shared__ float red[8];
  const int row = blockIdx.x;
  const float* xr = x + (size_t)row * W;
  float s = 0.f;
  for (int c = threadIdx.x; c < W; c += 256) { float v = xr[c]; s += v * v; }
  for (int d = 1; d < 32; d <<= 1) s += __shfl_xor(s, d, 32);
  int wave = threadIdx.x >> 5, lane = threadIdx.x & 31;
  if (lane == 0) red[wave] = s;
  __syncthreads();
  if (wave == 0) {
    float t = (lane < 8) ? red[lane] : 0.f;
    for (int d = 1; d < 8; d <<= 1) t += __shfl_xor(t, d, 32);
    if (lane == 0) red[0] = rsqrtf(t / (float)W + 1e-6f);
  }
  __syncthreads();
  float r = red[0];
  bf16_t* yr = y + (size_t)row * W;
  for (int c = threadIdx.x; c < W; c += 256) yr[c] = f2bf(xr[c] * r * w[c]);
}

__global__ void rope_kernel(const float* __restrict__ qkv, const float* __restrict__ cs,
                            const float* __restrict__ sn, const int* __restrict__ widx,
                            bf16_t* __restrict__ qg, bf16_t* __restrict__ kg,
                            bf16_t* __restrict__ vg)
{
  int i = blockIdx.x * 256 + threadIdx.x;     // NTOK*HID, exact multiple
  int t = i / HID, c = i % HID;
  int d = c % HD;
  int src = widx[t >> 2] * 4 + (t & 3);
  float cv = cs[src * HD + d], sv = sn[src * HD + d];
  const float* row = qkv + (size_t)t * QKV3;
  int cr = (d < 40) ? (c + 40) : (c - 40);
  float sg = (d < 40) ? -1.f : 1.f;
  qg[i] = f2bf(row[c] * cv + sg * row[cr] * sv);
  kg[i] = f2bf(row[HID + c] * cv + sg * row[HID + cr] * sv);
  vg[i] = f2bf(row[2 * HID + c]);
}

__global__ void silu_mul_kernel(const float* __restrict__ g, const float* __restrict__ u,
                                bf16_t* __restrict__ o)
{
  int i = blockIdx.x * 256 + threadIdx.x;     // NTOK*INTER, exact multiple
  float x = g[i];
  o[i] = f2bf((x / (1.f + __expf(-x))) * u[i]);
}

// ---------------------------------------------------------------------------
extern "C" void kernel_launch(void* const* d_in, const int* in_sizes, int n_in,
                              void* d_out, int out_size, void* d_ws, size_t ws_size,
                              hipStream_t stream)
{
  (void)in_sizes; (void)n_in; (void)out_size; (void)ws_size;
  const float* pixel  = (const float*)d_in[0];
  const float* cosb   = (const float*)d_in[1];
  const float* sinb   = (const float*)d_in[2];
  const int*   widx   = (const int*)d_in[3];
  const float* wpatch = (const float*)d_in[4];
  const float* norm1w = (const float*)d_in[5];
  const float* qkvw   = (const float*)d_in[6];
  const float* qkvb   = (const float*)d_in[7];
  const float* projw  = (const float*)d_in[8];
  const float* projb  = (const float*)d_in[9];
  const float* norm2w = (const float*)d_in[10];
  const float* gatew  = (const float*)d_in[11];
  const float* gateb  = (const float*)d_in[12];
  const float* upw    = (const float*)d_in[13];
  const float* upb    = (const float*)d_in[14];
  const float* downw  = (const float*)d_in[15];
  const float* downb  = (const float*)d_in[16];
  const float* mnormw = (const float*)d_in[17];
  const float* ml1w   = (const float*)d_in[18];
  const float* ml1b   = (const float*)d_in[19];
  const float* ml2w   = (const float*)d_in[20];
  const float* ml2b   = (const float*)d_in[21];
  const float* dnormw = (const float*)d_in[22];
  const float* dl1w   = (const float*)d_in[23];
  const float* dl1b   = (const float*)d_in[24];
  const float* dl2w   = (const float*)d_in[25];
  const float* dl2b   = (const float*)d_in[26];
  float* out = (float*)d_out;

  char* ws = (char*)d_ws;
  size_t off = 0;
  auto abf = [&](size_t e)->bf16_t* { bf16_t* p=(bf16_t*)(ws+off); off=(off+e*2+255)&~(size_t)255; return p; };
  auto af32= [&](size_t e)->float*  { float*  p=(float*) (ws+off); off=(off+e*4+255)&~(size_t)255; return p; };

  bf16_t* pxb      = abf((size_t)NTOK*PIX);
  bf16_t* wpatch_t = abf((size_t)PIX*HID);
  bf16_t* qkvw_t   = abf((size_t)6*HID*QKV3);
  bf16_t* projw_t  = abf((size_t)6*HID*HID);
  bf16_t* gatew_t  = abf((size_t)6*HID*INTER);
  bf16_t* upw_t    = abf((size_t)6*HID*INTER);
  bf16_t* downw_t  = abf((size_t)6*INTER*HID);
  bf16_t* ml1_t    = abf((size_t)UHID*UHID);
  bf16_t* ml2_t    = abf((size_t)UHID*ODIM);
  bf16_t* dl1_t    = abf((size_t)2*UHID*UHID);
  bf16_t* dl2_t    = abf((size_t)2*UHID*ODIM);
  float*  hbuf     = af32((size_t)NTOK*HID);
  bf16_t* xb       = abf((size_t)NTOK*HID);
  float*  qkvbuf   = af32((size_t)NTOK*QKV3);
  bf16_t* qg       = abf((size_t)NTOK*HID);
  bf16_t* kg       = abf((size_t)NTOK*HID);
  bf16_t* vg       = abf((size_t)NTOK*HID);
  bf16_t* ao       = abf((size_t)NTOK*HID);
  float*  gbuf     = af32((size_t)NTOK*INTER);
  float*  ubuf     = af32((size_t)NTOK*INTER);
  bf16_t* mgl      = abf((size_t)NTOK*INTER);
  bf16_t* xm       = abf((size_t)MROW*UHID);
  bf16_t* y1       = abf((size_t)MROW*UHID);

  auto cvt = [&](const float* w, bf16_t* wt, int K, int N){
    long long tot = (long long)K * N;
    cvtw_kernel<<<dim3((unsigned)((tot + 255) / 256)), dim3(256), 0, stream>>>(w, wt, K, N);
  };

  cvt(wpatch, wpatch_t, PIX, HID);
  for (int i = 0; i < 6; ++i) {
    cvt(qkvw  + (size_t)i*HID*QKV3,  qkvw_t  + (size_t)i*HID*QKV3,  HID, QKV3);
    cvt(projw + (size_t)i*HID*HID,   projw_t + (size_t)i*HID*HID,   HID, HID);
    cvt(gatew + (size_t)i*HID*INTER, gatew_t + (size_t)i*HID*INTER, HID, INTER);
    cvt(upw   + (size_t)i*HID*INTER, upw_t   + (size_t)i*HID*INTER, HID, INTER);
    cvt(downw + (size_t)i*INTER*HID, downw_t + (size_t)i*INTER*HID, INTER, HID);
  }
  cvt(ml1w, ml1_t, UHID, UHID);
  cvt(ml2w, ml2_t, UHID, ODIM);
  for (int j = 0; j < 2; ++j) {
    cvt(dl1w + (size_t)j*UHID*UHID, dl1_t + (size_t)j*UHID*UHID, UHID, UHID);
    cvt(dl2w + (size_t)j*UHID*ODIM, dl2_t + (size_t)j*UHID*ODIM, UHID, ODIM);
  }

  gather_px_kernel<<<(NTOK*PIX)/256, 256, 0, stream>>>(pixel, widx, pxb);
  gemm_bf16_wmma<0><<<dim3(HID/128, NTOK/128), 256, GEMM_SMEM, stream>>>(
      pxb, wpatch_t, nullptr, hbuf, nullptr, nullptr, NTOK, HID, PIX);

  for (int i = 0; i < 6; ++i) {
    rmsnorm_kernel<<<NTOK, 256, 0, stream>>>(hbuf, norm1w + (size_t)i*HID, xb, HID);
    gemm_bf16_wmma<0><<<dim3(QKV3/128, NTOK/128), 256, GEMM_SMEM, stream>>>(
        xb, qkvw_t + (size_t)i*HID*QKV3, qkvb + (size_t)i*QKV3,
        qkvbuf, nullptr, nullptr, NTOK, QKV3, HID);
    rope_kernel<<<(NTOK*HID)/256, 256, 0, stream>>>(qkvbuf, cosb, sinb, widx, qg, kg, vg);
    attn_wmma<<<NWIN*NHEAD, 128, 0, stream>>>(qg, kg, vg, ao);
    gemm_bf16_wmma<1><<<dim3(HID/128, NTOK/128), 256, GEMM_SMEM, stream>>>(
        ao, projw_t + (size_t)i*HID*HID, projb + (size_t)i*HID,
        hbuf, nullptr, nullptr, NTOK, HID, HID);
    rmsnorm_kernel<<<NTOK, 256, 0, stream>>>(hbuf, norm2w + (size_t)i*HID, xb, HID);
    gemm_bf16_wmma<0><<<dim3(INTER/128, NTOK/128), 256, GEMM_SMEM, stream>>>(
        xb, gatew_t + (size_t)i*HID*INTER, gateb + (size_t)i*INTER,
        gbuf, nullptr, nullptr, NTOK, INTER, HID);
    gemm_bf16_wmma<0><<<dim3(INTER/128, NTOK/128), 256, GEMM_SMEM, stream>>>(
        xb, upw_t + (size_t)i*HID*INTER, upb + (size_t)i*INTER,
        ubuf, nullptr, nullptr, NTOK, INTER, HID);
    silu_mul_kernel<<<(NTOK*INTER)/256, 256, 0, stream>>>(gbuf, ubuf, mgl);
    gemm_bf16_wmma<1><<<dim3(HID/128, NTOK/128), 256, GEMM_SMEM, stream>>>(
        mgl, downw_t + (size_t)i*INTER*HID, downb + (size_t)i*HID,
        hbuf, nullptr, nullptr, NTOK, HID, INTER);
    if (i == 2 || i == 4) {
      int j = (i == 2) ? 0 : 1;
      rmsnorm_kernel<<<MROW, 256, 0, stream>>>(hbuf, dnormw + (size_t)j*UHID, xm, UHID);
      gemm_bf16_wmma<2><<<dim3(UHID/128, MROW/128), 256, GEMM_SMEM, stream>>>(
          xm, dl1_t + (size_t)j*UHID*UHID, dl1b + (size_t)j*UHID,
          nullptr, y1, nullptr, MROW, UHID, UHID);
      gemm_bf16_wmma<3><<<dim3(ODIM/128, MROW/128), 256, GEMM_SMEM, stream>>>(
          y1, dl2_t + (size_t)j*UHID*ODIM, dl2b + (size_t)j*ODIM,
          out + (size_t)(1 + j)*MROW*ODIM, nullptr, widx, MROW, ODIM, UHID);
    }
  }
  rmsnorm_kernel<<<NTOK, 256, 0, stream>>>(hbuf, mnormw, xm, HID);
  gemm_bf16_wmma<2><<<dim3(UHID/128, MROW/128), 256, GEMM_SMEM, stream>>>(
      xm, ml1_t, ml1b, nullptr, y1, nullptr, MROW, UHID, UHID);
  gemm_bf16_wmma<3><<<dim3(ODIM/128, MROW/128), 256, GEMM_SMEM, stream>>>(
      y1, ml2_t, ml2b, out, nullptr, widx, MROW, ODIM, UHID);
}
